// MultiHeadAttention_86517821216329
// MI455X (gfx1250) — matrix operand order repaired and checked by reading, not compiled, not run on hardware
//
#include <hip/hip_runtime.h>
#include <hip/hip_bf16.h>

// B=8, L=1024, D=1024, H=16, DK=DV=64
#define HD  16
#define LQ  1024
#define DKV 64

#define ASYNC_STAGE 1   // use global_load_async_to_lds_b128 for bf16 tiles

typedef __attribute__((ext_vector_type(16))) __bf16      v16bf;
typedef __attribute__((ext_vector_type(8)))  float       v8f;
typedef __attribute__((ext_vector_type(4)))  unsigned int u32x4;

union BFrag { v16bf bf; u32x4 q[2]; };

__device__ __forceinline__ unsigned bfround(float f) {
  unsigned u = __float_as_uint(f);
  return u + 0x7FFFu + ((u >> 16) & 1u);   // rounded bf16 lives in bits[31:16]
}
// pack two rounded bf16 into one dword with a single v_perm_b32
__device__ __forceinline__ unsigned pk2(float a, float b) {
  return __builtin_amdgcn_perm(bfround(b), bfround(a), 0x07060302u);
}
__device__ __forceinline__ unsigned short f2bf(float f) {
  return (unsigned short)(bfround(f) >> 16);
}

#if ASYNC_STAGE
// Per-lane async copy of 16B global -> LDS (ASYNCcnt-tracked, no VGPR data).
__device__ __forceinline__ void async_g2l_b128(void* lds, const void* g) {
  unsigned            loff = (unsigned)(uintptr_t)lds;           // LDS byte offset
  unsigned long long  ga   = (unsigned long long)(uintptr_t)g;
  asm volatile("global_load_async_to_lds_b128 %0, %1, off"
               :: "v"(loff), "v"(ga) : "memory");
}
__device__ __forceinline__ void wait_async0() {
#if __has_builtin(__builtin_amdgcn_s_wait_asynccnt)
  __builtin_amdgcn_s_wait_asynccnt(0);
#else
  asm volatile("s_wait_asynccnt 0" ::: "memory");
#endif
}
#endif

// Generic bf16-WMMA GEMM, tile 128x128x32, 256 threads = 8 wave32 waves.
// AK:    0 = A is f32 row-major [M,K] (convert to bf16 on LDS store)
//        1 = A is bf16 row-major [M,K] (async copy to LDS)
// BKIND: 0 = B is f32 [K,N] row-major (transpose-stage to sB[n][k])
//        1 = B is bf16 [N,K] row-major (async copy; used for Q*K^T)
//        2 = B is bf16 [K,N] row-major (transpose-stage; used for attn*V)
// OM:    0 = head-split bf16 out + bias (projections)
//        1 = f32 out, *1/sqrt(64), mask -1e9 (scores)
//        2 = merge-heads bf16 out (attn*V)
//        3 = plain f32 out + bias (final projection)
template<int AK, int BKIND, int OM>
__global__ __launch_bounds__(256)
void gemm_wmma(const void* __restrict__ Ag, int lda, long long aStride,
               const void* __restrict__ Bg, int ldb, long long bStride,
               int Ksz, int Nsz,
               void* __restrict__ Cg, const float* __restrict__ bias,
               const int* __restrict__ mask)
{
  constexpr int BM = 128, BN = 128, BKT = 32, LS = BKT + 8; // pad 8 shorts
  __shared__ __align__(16) unsigned short sA[BM * LS];
  __shared__ __align__(16) unsigned short sB[BN * LS];

  const int tid = threadIdx.x;
  const int z   = blockIdx.z;
  const int m0  = blockIdx.y * BM;
  const int n0  = blockIdx.x * BN;

  const float*          Af = (const float*)Ag + (size_t)z * aStride;
  const unsigned short* Ab = (const unsigned short*)Ag + (size_t)z * aStride;
  const float*          Bf = (const float*)Bg + (size_t)z * bStride;
  const unsigned short* Bb = (const unsigned short*)Bg + (size_t)z * bStride;

  const int lane = tid & 31;
  const int l16  = lane & 15;
  const int lh   = lane >> 4;       // half-wave select
  const int wid  = tid >> 5;
  const int wm   = wid >> 1;        // 0..3  -> 32-row band
  const int wn   = wid & 1;         // 0..1  -> 64-col band

  v8f acc[2][4];
  #pragma unroll
  for (int mi = 0; mi < 2; ++mi)
    #pragma unroll
    for (int ni = 0; ni < 4; ++ni)
      acc[mi][ni] = v8f{0.f,0.f,0.f,0.f,0.f,0.f,0.f,0.f};

  for (int k0 = 0; k0 < Ksz; k0 += BKT) {
    __syncthreads();

    // ---------- stage A tile (128x32 bf16) ----------
    {
      const int row = tid >> 1;          // 0..127
      const int cg  = (tid & 1) * 16;    // 0 or 16
      unsigned short* dst = &sA[row * LS + cg];
      if constexpr (AK == 0) {
        const float* ap = Af + (size_t)(m0 + row) * lda + k0 + cg;
        if (k0 + BKT < Ksz) __builtin_prefetch(ap + BKT, 0, 0);
        float4 f0 = *(const float4*)(ap + 0);
        float4 f1 = *(const float4*)(ap + 4);
        float4 f2 = *(const float4*)(ap + 8);
        float4 f3 = *(const float4*)(ap + 12);
        u32x4 u0 = { pk2(f0.x,f0.y), pk2(f0.z,f0.w), pk2(f1.x,f1.y), pk2(f1.z,f1.w) };
        u32x4 u1 = { pk2(f2.x,f2.y), pk2(f2.z,f2.w), pk2(f3.x,f3.y), pk2(f3.z,f3.w) };
        *(u32x4*)(dst)     = u0;
        *(u32x4*)(dst + 8) = u1;
      } else {
        const unsigned short* ap = Ab + (size_t)(m0 + row) * lda + k0 + cg;
#if ASYNC_STAGE
        async_g2l_b128(dst,     ap);
        async_g2l_b128(dst + 8, ap + 8);
#else
        *(u32x4*)(dst)     = *(const u32x4*)(ap);
        *(u32x4*)(dst + 8) = *(const u32x4*)(ap + 8);
#endif
      }
    }

    // ---------- stage B tile into sB[n][k] (128x32 bf16) ----------
    if constexpr (BKIND == 0) {          // f32 [K][N] -> transpose-stage
      if (k0 + BKT < Ksz)
        __builtin_prefetch(&Bf[(size_t)(k0 + BKT + (tid >> 7)) * ldb + n0 + (tid & 127)], 0, 0);
      #pragma unroll
      for (int j = 0; j < 16; ++j) {
        int e = tid + j * 256;
        int n = e & (BN - 1), k = e >> 7;
        float f = Bf[(size_t)(k0 + k) * ldb + n0 + n];
        sB[n * LS + k] = f2bf(f);
      }
    } else if constexpr (BKIND == 1) {   // bf16 [N][K] -> direct rows
      const int n  = tid >> 1;
      const int cg = (tid & 1) * 16;
      const unsigned short* bp = Bb + (size_t)(n0 + n) * ldb + k0 + cg;
      unsigned short* dst = &sB[n * LS + cg];
#if ASYNC_STAGE
      async_g2l_b128(dst,     bp);
      async_g2l_b128(dst + 8, bp + 8);
#else
      *(u32x4*)(dst)     = *(const u32x4*)(bp);
      *(u32x4*)(dst + 8) = *(const u32x4*)(bp + 8);
#endif
    } else {                             // bf16 [K][N] -> transpose-stage
      if (k0 + BKT < Ksz)
        __builtin_prefetch(&Bb[(size_t)(k0 + BKT + (tid >> 7)) * ldb], 0, 0);
      #pragma unroll
      for (int j = 0; j < 16; ++j) {
        int e = tid + j * 256;
        int n = e & (BN - 1), k = e >> 7;
        unsigned short val = 0;
        if (n0 + n < Nsz) val = Bb[(size_t)(k0 + k) * ldb + n0 + n];
        sB[n * LS + k] = val;
      }
    }

#if ASYNC_STAGE
    if constexpr (AK == 1 || BKIND == 1) wait_async0();
#endif
    __syncthreads();

    // ---------- load fragments (ISA §7.12.2 layouts) + WMMA ----------
    BFrag a[2], b[4];
    #pragma unroll
    for (int mi = 0; mi < 2; ++mi) {
      // A 16x32: lanes 0-15 row m, elems = K[h*8..+8) and K[16+h*8..+8)
      const unsigned short* p = &sA[(wm * 32 + mi * 16 + l16) * LS];
      a[mi].q[0] = *(const u32x4*)(p + lh * 8);
      a[mi].q[1] = *(const u32x4*)(p + 16 + lh * 8);
    }
    #pragma unroll
    for (int ni = 0; ni < 4; ++ni) {
      // B 32x16: lane holds col, elems = K[h*8..+8) and K[16+h*8..+8)
      const unsigned short* p = &sB[(wn * 64 + ni * 16 + l16) * LS + lh * 8];
      b[ni].q[0] = *(const u32x4*)(p);
      b[ni].q[1] = *(const u32x4*)(p + 16);
    }
    #pragma unroll
    for (int mi = 0; mi < 2; ++mi)
      #pragma unroll
      for (int ni = 0; ni < 4; ++ni)
        acc[mi][ni] = __builtin_amdgcn_wmma_f32_16x16x32_bf16(
            false, a[mi].bf, false, b[ni].bf, (short)0, acc[mi][ni], false, false);
  }

  // ---------- epilogue (C layout: col = lane%16, row = (lane/16)*8 + r) ----------
  #pragma unroll
  for (int mi = 0; mi < 2; ++mi) {
    #pragma unroll
    for (int ni = 0; ni < 4; ++ni) {
      #pragma unroll
      for (int r = 0; r < 8; ++r) {
        int row = m0 + wm * 32 + mi * 16 + lh * 8 + r;
        int col = n0 + wn * 64 + ni * 16 + l16;
        if (col >= Nsz) continue;
        float val = acc[mi][ni][r];
        if constexpr (OM == 0) {            // head-split bf16: [B,H,L,64]
          int b_ = row >> 10, l_ = row & 1023;
          int h_ = col >> 6,  d_ = col & 63;
          val += bias[col];
          ((unsigned short*)Cg)[(((size_t)(b_ * HD + h_)) * LQ + l_) * DKV + d_] = f2bf(val);
        } else if constexpr (OM == 1) {     // scores: scale, mask, f32
          int b_ = z >> 4;
          val *= 0.125f;                    // 1/sqrt(64)
          size_t midx = (size_t)b_ * LQ * LQ + (size_t)row * LQ + col;
          if (mask[midx] == 0) val = -1e9f;
          ((float*)Cg)[(size_t)z * LQ * LQ + (size_t)row * LQ + col] = val;
        } else if constexpr (OM == 2) {     // merge-heads bf16: [B,L,H*DV]
          int b_ = z >> 4, h_ = z & 15;
          ((unsigned short*)Cg)[((size_t)(b_ * LQ + row)) * (HD * DKV) + h_ * DKV + col] = f2bf(val);
        } else {                            // plain f32 + bias
          ((float*)Cg)[(size_t)row * Nsz + col] = val + bias[col];
        }
      }
    }
  }
}

// Softmax over the QUERY axis (axis=2 of [B,H,Lq,Lk]) — column softmax.
// One thread per k-column; coalesced (adjacent threads -> adjacent k).
__global__ __launch_bounds__(256)
void softmax_qaxis(float* __restrict__ attn)
{
  const int k = blockIdx.x * 256 + threadIdx.x;
  float* p = attn + (size_t)blockIdx.y * LQ * LQ + k;
  float m = -3.0e38f, s = 0.f;
  for (int q = 0; q < LQ; ++q) {
    float x  = p[(size_t)q * LQ];
    float nm = fmaxf(m, x);
    s = s * __expf(m - nm) + __expf(x - nm);
    m = nm;
  }
  const float inv = 1.0f / s;
  for (int q = 0; q < LQ; ++q) {
    size_t off = (size_t)q * LQ;
    p[off] = __expf(p[off] - m) * inv;
  }
}

extern "C" void kernel_launch(void* const* d_in, const int* in_sizes, int n_in,
                              void* d_out, int out_size, void* d_ws, size_t ws_size,
                              hipStream_t stream)
{
  const float* q    = (const float*)d_in[0];
  const float* k    = (const float*)d_in[1];
  const float* v    = (const float*)d_in[2];
  const int*   mask = (const int*)d_in[3];
  const float* Wq   = (const float*)d_in[4];
  const float* bq   = (const float*)d_in[5];
  const float* Wk   = (const float*)d_in[6];
  const float* bk   = (const float*)d_in[7];
  const float* Wv   = (const float*)d_in[8];
  const float* bv   = (const float*)d_in[9];
  const float* Wf   = (const float*)d_in[10];
  const float* bf   = (const float*)d_in[11];

  const size_t PROJ = (size_t)8 * HD * LQ * DKV;   // 8,388,608 elements

  unsigned short* qh = (unsigned short*)d_ws;      // [B,H,L,64] bf16
  unsigned short* kh = qh + PROJ;
  unsigned short* vh = kh + PROJ;
  unsigned short* oh = vh + PROJ;                  // [B,L,H*64] bf16

  float* out  = (float*)d_out;                     // (B,L,1024) f32
  float* attn = out + (size_t)8 * LQ * 1024;       // (B,H,L,L)  f32

  dim3 blk(256);

  // Q/K/V projections: [8192,1024] x [1024,1024] -> head-split bf16
  gemm_wmma<0,0,0><<<dim3(8, 64, 1), blk, 0, stream>>>(q, 1024, 0, Wq, 1024, 0,
                                                       1024, 1024, qh, bq, nullptr);
  gemm_wmma<0,0,0><<<dim3(8, 64, 1), blk, 0, stream>>>(k, 1024, 0, Wk, 1024, 0,
                                                       1024, 1024, kh, bk, nullptr);
  gemm_wmma<0,0,0><<<dim3(8, 64, 1), blk, 0, stream>>>(v, 1024, 0, Wv, 1024, 0,
                                                       1024, 1024, vh, bv, nullptr);

  // scores: per (b,h) [1024,64] x [64,1024] -> masked scaled f32 attn
  gemm_wmma<1,1,1><<<dim3(8, 8, 128), blk, 0, stream>>>(qh, 64, 65536, kh, 64, 65536,
                                                        64, 1024, attn, nullptr, mask);

  // softmax over query axis (in place on attn)
  softmax_qaxis<<<dim3(4, 128), blk, 0, stream>>>(attn);

  // attn @ V: per (b,h) [1024,1024] x [1024,64] -> merge-heads bf16
  gemm_wmma<0,2,2><<<dim3(1, 8, 128), blk, 0, stream>>>(attn, 1024, (long long)LQ * LQ,
                                                        vh, 64, 65536,
                                                        1024, 64, oh, nullptr, nullptr);

  // final projection: [8192,1024] x [1024,1024] + bias -> f32 out
  gemm_wmma<1,0,3><<<dim3(8, 64, 1), blk, 0, stream>>>(oh, 1024, 0, Wf, 1024, 0,
                                                       1024, 1024, out, bf, nullptr);
}
